// DockingTimeModel_66297115181624
// MI455X (gfx1250) — compile-verified, matrix operand order checked by reading, not run on hardware
//
#include <hip/hip_runtime.h>
#include <hip/hip_bf16.h>

// ---------------- WMMA bf16 helpers (gfx1250, wave32) ----------------
typedef __attribute__((ext_vector_type(16))) __bf16 v16bf;
typedef __attribute__((ext_vector_type(8)))  float  v8f;

union FragU { v16bf v; unsigned short u[16]; uint4 q[2]; };

// Hardware bf16 convert (v_cvt_*bf16_f32), RNE.
__device__ __forceinline__ unsigned short f2bf_u(float f) {
    __bf16 h = (__bf16)f;
    unsigned short s;
    __builtin_memcpy(&s, &h, 2);
    return s;
}

// Fragment K pattern (ISA 7.12.2, 16-bit A/B, wave32):
//   element e, hi = lane>=16  ->  k = (e<8 ? 0 : 16) + hi*8 + (e&7)
// => each 8-element half is K-contiguous: half0 at k0+hi*8, half1 at k0+16+hi*8.

// A fragment of (a + b) elementwise (x + agg), row-major f32, ld multiple of 32.
__device__ __forceinline__ v16bf loadA_sum_f32(const float* __restrict__ a,
                                               const float* __restrict__ b,
                                               long row0, int ld, int k0, int lane) {
    const int m  = lane & 15;
    const int hi = (lane >> 4) & 1;
    const long base = (row0 + m) * (long)ld + k0 + hi * 8;
    float4 a0 = *(const float4*)(a + base);
    float4 a1 = *(const float4*)(a + base + 4);
    float4 a2 = *(const float4*)(a + base + 16);
    float4 a3 = *(const float4*)(a + base + 20);
    float4 c0 = *(const float4*)(b + base);
    float4 c1 = *(const float4*)(b + base + 4);
    float4 c2 = *(const float4*)(b + base + 16);
    float4 c3 = *(const float4*)(b + base + 20);
    FragU f;
    f.u[0]  = f2bf_u(a0.x + c0.x); f.u[1]  = f2bf_u(a0.y + c0.y);
    f.u[2]  = f2bf_u(a0.z + c0.z); f.u[3]  = f2bf_u(a0.w + c0.w);
    f.u[4]  = f2bf_u(a1.x + c1.x); f.u[5]  = f2bf_u(a1.y + c1.y);
    f.u[6]  = f2bf_u(a1.z + c1.z); f.u[7]  = f2bf_u(a1.w + c1.w);
    f.u[8]  = f2bf_u(a2.x + c2.x); f.u[9]  = f2bf_u(a2.y + c2.y);
    f.u[10] = f2bf_u(a2.z + c2.z); f.u[11] = f2bf_u(a2.w + c2.w);
    f.u[12] = f2bf_u(a3.x + c3.x); f.u[13] = f2bf_u(a3.y + c3.y);
    f.u[14] = f2bf_u(a3.z + c3.z); f.u[15] = f2bf_u(a3.w + c3.w);
    return f.v;
}

// Edge A fragment: 16 rows x K=16 (zero-padded to 32), row-major f32, ld=16.
__device__ __forceinline__ v16bf loadA_edge(const float* __restrict__ ea, long row0, int lane) {
    const int m  = lane & 15;
    const int hi = (lane >> 4) & 1;
    const float* r = ea + (row0 + m) * 16 + hi * 8;
    float4 x0 = *(const float4*)(r);
    float4 x1 = *(const float4*)(r + 4);
    FragU f;
    f.u[0] = f2bf_u(x0.x); f.u[1] = f2bf_u(x0.y); f.u[2] = f2bf_u(x0.z); f.u[3] = f2bf_u(x0.w);
    f.u[4] = f2bf_u(x1.x); f.u[5] = f2bf_u(x1.y); f.u[6] = f2bf_u(x1.z); f.u[7] = f2bf_u(x1.w);
    f.q[1] = make_uint4(0u, 0u, 0u, 0u);   // K=16..31 zero pad
    return f.v;
}

// B fragment from pre-transposed bf16 weight Wt[n][k] (row-major by n, K columns).
// Two 16-byte loads, no conversion.
__device__ __forceinline__ v16bf loadB_t(const unsigned short* __restrict__ Wt,
                                         int K, int n0, int k0, int lane) {
    const int n  = n0 + (lane & 15);
    const int hi = (lane >> 4) & 1;
    const unsigned short* p = Wt + (long)n * K + k0 + hi * 8;
    FragU f;
    f.q[0] = *(const uint4*)(p);
    f.q[1] = *(const uint4*)(p + 16);
    return f.v;
}

// Edge B fragment: Wt[n][k] with K=16 (zero-pad K=16..31).
__device__ __forceinline__ v16bf loadB_edge(const unsigned short* __restrict__ Wt,
                                            int n0, int lane) {
    const int n  = n0 + (lane & 15);
    const int hi = (lane >> 4) & 1;
    FragU f;
    f.q[0] = *(const uint4*)(Wt + n * 16 + hi * 8);
    f.q[1] = make_uint4(0u, 0u, 0u, 0u);
    return f.v;
}

// A fragment from a per-wave LDS tile of bf16 bits, row-major [16][ldk], 16B aligned.
__device__ __forceinline__ v16bf loadA_lds(const unsigned short* t, int ldk, int k0, int lane) {
    const int m  = lane & 15;
    const int hi = (lane >> 4) & 1;
    const unsigned short* p = t + m * ldk + k0 + hi * 8;
    FragU f;
    f.q[0] = *(const uint4*)(p);
    f.q[1] = *(const uint4*)(p + 16);
    return f.v;
}

// ---------------- Kernel 0: weight prep  Wt[n][k] (bf16) = W[k][n] (f32) ----------------
__global__ void wt_bf16_kernel(const float* __restrict__ W, unsigned short* __restrict__ Wt,
                               int K, int Nc) {
    int i = blockIdx.x * blockDim.x + threadIdx.x;
    if (i >= K * Nc) return;
    int n = i / K, k = i - n * K;
    Wt[i] = f2bf_u(W[k * Nc + n]);
}

// ---------------- Kernel 1: edge linear  elin = edge_attr @ W(16x64) + b ----------------
__global__ void edge_lin_kernel(const float* __restrict__ ea, const unsigned short* __restrict__ Wt,
                                const float* __restrict__ bias, float* __restrict__ out,
                                int nedges) {
    const int wave = threadIdx.x >> 5;
    const int lane = threadIdx.x & 31;
    const long row0 = ((long)blockIdx.x * 8 + wave) * 16;
    if (row0 >= nedges) return;                 // wave-uniform: EXEC all-ones at WMMA
    const int hi = (lane >> 4) & 1;
    const int nn = lane & 15;

    v16bf a = loadA_edge(ea, row0, lane);
#pragma unroll
    for (int j = 0; j < 4; ++j) {
        v16bf b = loadB_edge(Wt, j * 16, lane);
        v8f acc = {};
        acc = __builtin_amdgcn_wmma_f32_16x16x32_bf16(false, a, false, b, (short)0, acc, false, false);
        const int col = j * 16 + nn;
        const float bv = bias[col];
#pragma unroll
        for (int r = 0; r < 8; ++r)
            out[(row0 + r + 8 * hi) * 64 + col] = acc[r] + bv;
    }
}

// ---------------- Kernel 2: message + scatter  agg[dst] += relu(x[src] + elin) ----------------
__global__ void scatter_kernel(const float* __restrict__ xfeat, const float* __restrict__ elin,
                               const long long* __restrict__ ei, float* __restrict__ agg,
                               int nedges) {
    const long tid = (long)blockIdx.x * blockDim.x + threadIdx.x;
    const long e   = tid >> 2;
    const int part = (int)(tid & 3);
    if (e >= nedges) return;
    const long s = ei[e];
    const long d = ei[(long)nedges + e];
    const float4* xs = (const float4*)(xfeat + s * 64) + part * 4;
    const float4* el = (const float4*)(elin  + e * 64) + part * 4;
    float* ag = agg + d * 64 + part * 16;
#pragma unroll
    for (int q = 0; q < 4; ++q) {
        float4 xv = xs[q];
        float4 ev = el[q];
        float m0 = xv.x + ev.x; m0 = m0 > 0.f ? m0 : 0.f;
        float m1 = xv.y + ev.y; m1 = m1 > 0.f ? m1 : 0.f;
        float m2 = xv.z + ev.z; m2 = m2 > 0.f ? m2 : 0.f;
        float m3 = xv.w + ev.w; m3 = m3 > 0.f ? m3 : 0.f;
        unsafeAtomicAdd(ag + q * 4 + 0, m0);
        unsafeAtomicAdd(ag + q * 4 + 1, m1);
        unsafeAtomicAdd(ag + q * 4 + 2, m2);
        unsafeAtomicAdd(ag + q * 4 + 3, m3);
    }
}

// ---------------- Kernel 3: fused node MLP  out = [relu]( relu((x+agg)@W1+b1) @ W2 + b2 ) ----
template <int IN, int HID, int OUT, bool RELU_OUT>
__global__ void node_mlp_kernel(const float* __restrict__ xin, const float* __restrict__ agg,
                                const unsigned short* __restrict__ Wt1, const float* __restrict__ b1,
                                const unsigned short* __restrict__ Wt2, const float* __restrict__ b2,
                                float* __restrict__ out, int nrows) {
    __shared__ __align__(16) unsigned short t_lds[8 * 16 * HID];  // per-wave 16xHID bf16 staging
    const int wave = threadIdx.x >> 5;
    const int lane = threadIdx.x & 31;
    const long row0 = ((long)blockIdx.x * 8 + wave) * 16;
    if (row0 >= nrows) return;                  // wave-uniform exit; no block barriers used
    unsigned short* t = t_lds + wave * 16 * HID;
    const int hi = (lane >> 4) & 1;
    const int nn = lane & 15;

    // Hoisted A fragments of hin = x + agg
    v16bf afrag[IN / 32];
#pragma unroll
    for (int c = 0; c < IN / 32; ++c)
        afrag[c] = loadA_sum_f32(xin, agg, row0, IN, c * 32, lane);

    // GEMM1 + bias + relu -> LDS (bf16), row-major so it reloads as A fragments
#pragma unroll
    for (int j = 0; j < HID / 16; ++j) {
        v8f acc = {};
#pragma unroll
        for (int c = 0; c < IN / 32; ++c) {
            v16bf b = loadB_t(Wt1, IN, j * 16, c * 32, lane);
            acc = __builtin_amdgcn_wmma_f32_16x16x32_bf16(false, afrag[c], false, b, (short)0, acc, false, false);
        }
        const int col = j * 16 + nn;
        const float bias = b1[col];
#pragma unroll
        for (int r = 0; r < 8; ++r) {
            float v = acc[r] + bias;
            v = v > 0.f ? v : 0.f;
            t[(r + 8 * hi) * HID + col] = f2bf_u(v);  // same-wave LDS: in-order, no barrier
        }
    }

    // GEMM2: out = t @ W2 + b2
    v16bf afrag2[HID / 32];
#pragma unroll
    for (int c = 0; c < HID / 32; ++c)
        afrag2[c] = loadA_lds(t, HID, c * 32, lane);

#pragma unroll
    for (int j = 0; j < OUT / 16; ++j) {
        v8f acc = {};
#pragma unroll
        for (int c = 0; c < HID / 32; ++c) {
            v16bf b = loadB_t(Wt2, HID, j * 16, c * 32, lane);
            acc = __builtin_amdgcn_wmma_f32_16x16x32_bf16(false, afrag2[c], false, b, (short)0, acc, false, false);
        }
        const int col = j * 16 + nn;
        const float bias = b2[col];
#pragma unroll
        for (int r = 0; r < 8; ++r) {
            float v = acc[r] + bias;
            if (RELU_OUT) v = v > 0.f ? v : 0.f;
            out[(row0 + r + 8 * hi) * OUT + col] = v;
        }
    }
}

// ---------------- Kernel 4: mean-pool accumulate ----------------
__global__ void pool_kernel(const float* __restrict__ h, const long long* __restrict__ batch,
                            float* __restrict__ sums, float* __restrict__ cnt, int nnodes) {
    const long tid = (long)blockIdx.x * blockDim.x + threadIdx.x;
    const long n   = tid >> 2;
    const int part = (int)(tid & 3);
    if (n >= nnodes) return;
    const long g = batch[n];
    const float* hp = h + n * 128 + part * 32;
    float* sp = sums + g * 128 + part * 32;
#pragma unroll
    for (int q = 0; q < 32; ++q) unsafeAtomicAdd(sp + q, hp[q]);
    if (part == 0) unsafeAtomicAdd(cnt + g, 1.0f);
}

// ---------------- Kernel 5: per-graph head MLP (140->128->64->32->16->1) ----------------
__global__ void head_kernel(const float* __restrict__ sums, const float* __restrict__ cnt,
                            const float* __restrict__ usr,
                            const float* w1, const float* bb1, const float* w2, const float* bb2,
                            const float* w3, const float* bb3, const float* w4, const float* bb4,
                            const float* w5, const float* bb5, float* __restrict__ out) {
    __shared__ float za[140];
    __shared__ float zb[128];
    const int g = blockIdx.x;
    const int t = threadIdx.x;   // 128 threads
    float c = cnt[g]; c = c > 1.f ? c : 1.f;
    za[t] = sums[g * 128 + t] / c;
    if (t < 12) za[128 + t] = usr[g * 12 + t];
    __syncthreads();
    { float a = bb1[t]; for (int k = 0; k < 140; ++k) a += za[k] * w1[k * 128 + t]; zb[t] = a > 0.f ? a : 0.f; }
    __syncthreads();
    if (t < 64) { float a = bb2[t]; for (int k = 0; k < 128; ++k) a += zb[k] * w2[k * 64 + t]; za[t] = a > 0.f ? a : 0.f; }
    __syncthreads();
    if (t < 32) { float a = bb3[t]; for (int k = 0; k < 64; ++k) a += za[k] * w3[k * 32 + t]; zb[t] = a > 0.f ? a : 0.f; }
    __syncthreads();
    if (t < 16) { float a = bb4[t]; for (int k = 0; k < 32; ++k) a += zb[k] * w4[k * 16 + t]; za[t] = a > 0.f ? a : 0.f; }
    __syncthreads();
    if (t == 0) { float a = bb5[0]; for (int k = 0; k < 16; ++k) a += za[k] * w5[k]; out[g] = a; }
}

// ---------------- Host orchestration ----------------
extern "C" void kernel_launch(void* const* d_in, const int* in_sizes, int n_in,
                              void* d_out, int out_size, void* d_ws, size_t ws_size,
                              hipStream_t stream) {
    const float*     x     = (const float*)d_in[0];
    const long long* ei    = (const long long*)d_in[1];
    const float*     ea    = (const float*)d_in[2];
    const long long* batch = (const long long*)d_in[3];
    const float*     usr   = (const float*)d_in[4];
    const float* e1_w = (const float*)d_in[5];  const float* e1_b = (const float*)d_in[6];
    const float* n1w1 = (const float*)d_in[7];  const float* n1b1 = (const float*)d_in[8];
    const float* n1w2 = (const float*)d_in[9];  const float* n1b2 = (const float*)d_in[10];
    const float* e2_w = (const float*)d_in[11]; const float* e2_b = (const float*)d_in[12];
    const float* n2w1 = (const float*)d_in[13]; const float* n2b1 = (const float*)d_in[14];
    const float* n2w2 = (const float*)d_in[15]; const float* n2b2 = (const float*)d_in[16];
    const float* h1w  = (const float*)d_in[17]; const float* h1b  = (const float*)d_in[18];
    const float* h2w  = (const float*)d_in[19]; const float* h2b  = (const float*)d_in[20];
    const float* h3w  = (const float*)d_in[21]; const float* h3b  = (const float*)d_in[22];
    const float* h4w  = (const float*)d_in[23]; const float* h4b  = (const float*)d_in[24];
    const float* h5w  = (const float*)d_in[25]; const float* h5b  = (const float*)d_in[26];
    float* out = (float*)d_out;

    constexpr long N_ = 200000, E_ = 400000, G_ = 4096;

    float* ws   = (float*)d_ws;
    float* elin = ws;                  // E*64
    float* agg  = elin + E_ * 64;      // N*64
    float* h1   = agg  + N_ * 64;      // N*64
    float* h2   = h1   + N_ * 64;      // N*128
    float* sums = h2   + N_ * 128;     // G*128
    float* cnt  = sums + G_ * 128;     // G (contiguous with sums)

    // bf16 transposed weights (ushort); all offsets stay 16B-aligned
    unsigned short* wb     = (unsigned short*)(cnt + G_);
    unsigned short* wt_e1  = wb;                  // 64*16
    unsigned short* wt_11  = wt_e1 + 64 * 16;     // 64*64
    unsigned short* wt_12  = wt_11 + 64 * 64;     // 64*64
    unsigned short* wt_e2  = wt_12 + 64 * 64;     // 64*16
    unsigned short* wt_21  = wt_e2 + 64 * 16;     // 128*64
    unsigned short* wt_22  = wt_21 + 128 * 64;    // 128*128

    const int EB = (int)((E_ / 16 + 7) / 8);    // 16-edge tiles, 8 waves/block
    const int NB = (int)((N_ / 16 + 7) / 8);    // 16-node tiles, 8 waves/block
    const int SB = (int)((E_ * 4 + 255) / 256); // scatter: 4 threads/edge
    const int PB = (int)((N_ * 4 + 255) / 256); // pool: 4 threads/node

    // ---- weight prep (tiny) ----
    wt_bf16_kernel<<<(16 * 64 + 255) / 256, 256, 0, stream>>>(e1_w, wt_e1, 16, 64);
    wt_bf16_kernel<<<(64 * 64 + 255) / 256, 256, 0, stream>>>(n1w1, wt_11, 64, 64);
    wt_bf16_kernel<<<(64 * 64 + 255) / 256, 256, 0, stream>>>(n1w2, wt_12, 64, 64);
    wt_bf16_kernel<<<(16 * 64 + 255) / 256, 256, 0, stream>>>(e2_w, wt_e2, 16, 64);
    wt_bf16_kernel<<<(64 * 128 + 255) / 256, 256, 0, stream>>>(n2w1, wt_21, 64, 128);
    wt_bf16_kernel<<<(128 * 128 + 255) / 256, 256, 0, stream>>>(n2w2, wt_22, 128, 128);

    // ---- GINE layer 1 ----
    edge_lin_kernel<<<EB, 256, 0, stream>>>(ea, wt_e1, e1_b, elin, (int)E_);
    hipMemsetAsync(agg, 0, N_ * 64 * sizeof(float), stream);
    scatter_kernel<<<SB, 256, 0, stream>>>(x, elin, ei, agg, (int)E_);
    node_mlp_kernel<64, 64, 64, true><<<NB, 256, 0, stream>>>(x, agg, wt_11, n1b1, wt_12, n1b2, h1, (int)N_);

    // ---- GINE layer 2 ----
    edge_lin_kernel<<<EB, 256, 0, stream>>>(ea, wt_e2, e2_b, elin, (int)E_);
    hipMemsetAsync(agg, 0, N_ * 64 * sizeof(float), stream);
    scatter_kernel<<<SB, 256, 0, stream>>>(h1, elin, ei, agg, (int)E_);
    node_mlp_kernel<64, 128, 128, false><<<NB, 256, 0, stream>>>(h1, agg, wt_21, n2b1, wt_22, n2b2, h2, (int)N_);

    // ---- global mean pool + head MLP ----
    hipMemsetAsync(sums, 0, (G_ * 128 + G_) * sizeof(float), stream);
    pool_kernel<<<PB, 256, 0, stream>>>(h2, batch, sums, cnt, (int)N_);
    head_kernel<<<(int)G_, 128, 0, stream>>>(sums, cnt, usr,
                                             h1w, h1b, h2w, h2b, h3w, h3b, h4w, h4b, h5w, h5b, out);
}
